// GINEClassifier_56221121904766
// MI455X (gfx1250) — matrix operand order repaired and checked by reading, not compiled, not run on hardware
//
#include <hip/hip_runtime.h>
#include <hip/hip_bf16.h>
#include <math.h>

// ---------------- problem constants ----------------
#define B_   32
#define N_   4096
#define E_   16384
#define DN_  34
#define H_   128
#define L_   5
#define JK_  768          // H*(L+1)
#define HC_  193
#define FUS_ 256
#define NC_  9
#define BN_INV 0.9999950000374997f   // 1/sqrt(1+1e-5)

typedef __attribute__((ext_vector_type(16))) _Float16 v16h;
typedef __attribute__((ext_vector_type(8)))  _Float16 v8h;
typedef __attribute__((ext_vector_type(8)))  float    v8f;

__device__ __forceinline__ float fast_tanh(float x) {
    // 1 - 2/(exp(2x)+1): branch-free, saturates to +/-1, no NaN for large |x|
    float ex = __expf(2.0f * x);
    return fmaf(-2.0f, __frcp_rn(ex + 1.0f), 1.0f);
}

// ============================================================
// Weight pack: W[K x 128] f32 (row-major) -> f16 B-fragments in WMMA order.
// Packed layout: P[tile(8)][kblk(K/32)][lane(32)][e(16)]  (32B per lane chunk)
//   k = kblk*32 + (e>>3)*16 + (lane>>4)*8 + (e&7) ; col = tile*16 + (lane&15)
// ============================================================
template<int K>
__global__ void pack_w_kernel(const float* __restrict__ W, _Float16* __restrict__ P)
{
    int idx = blockIdx.x * 256 + threadIdx.x;     // [0, K*128)
    int col = idx & 127, k = idx >> 7;
    int tile = col >> 4, lane15 = col & 15;
    int kblk = k >> 5, laneHalf = (k >> 3) & 1, e = (k & 7) + (((k >> 4) & 1) << 3);
    int lane = laneHalf * 16 + lane15;
    P[(((size_t)tile * (K / 32) + kblk) * 32 + lane) * 16 + e] = (_Float16)W[idx];
}

// ============================================================
// WMMA GEMM:  Y[M x 128] = epilogue(X[M x K] @ W[K x 128] + bias)
// MODE 0: bias only | 1: bias+BN+ReLU | 3: bias+tanh
// 8 waves along M; each wave computes 16 rows x 64 cols (4 independent
// 16x16 WMMA accumulators -> hides WMMA->WMMA RAW hazard window).
// Block tile 128x64 ; grid = (M/128, 2).  K compile-time, fully unrolled.
// B from pre-packed f16 fragments (lane-contiguous b128 loads).
// ============================================================
template<int K, int MODE, bool INF16, bool OUTF16>
__global__ __launch_bounds__(256) void wmma_gemm(
    const void* __restrict__ Xv, const _Float16* __restrict__ PB,
    const float* __restrict__ bias,
    const float* __restrict__ bn_g, const float* __restrict__ bn_b,
    void* __restrict__ Yv)
{
    const int lane     = threadIdx.x & 31;
    const int wave     = threadIdx.x >> 5;           // 0..7, all along M
    const int rowBase  = blockIdx.x * 128 + wave * 16;
    const int colBase  = blockIdx.y * 64;            // four 16-col tiles
    const int laneHalf = lane >> 4;
    const int lane15   = lane & 15;
    const int tile0    = colBase >> 4;

    v8f acc[4] = {};

    const float*    xrowF = (const float*)Xv    + (size_t)(rowBase + lane15) * (size_t)K;
    const _Float16* xrowH = (const _Float16*)Xv + (size_t)(rowBase + lane15) * (size_t)K;

    #pragma unroll
    for (int k0 = 0; k0 < K; k0 += 32) {
        const int kblk = k0 >> 5;
        // ---- A fragment (16x32 f16): lane needs K = k0+laneHalf*8+0..7 and
        // K = k0+16+laneHalf*8+0..7 -> two contiguous 8-element runs.
        v16h a;
        if (INF16) {
            v8h lo = *(const v8h*)(xrowH + k0 + laneHalf * 8);
            v8h hi = *(const v8h*)(xrowH + k0 + 16 + laneHalf * 8);
            #pragma unroll
            for (int e = 0; e < 8; ++e) { a[e] = lo[e]; a[8 + e] = hi[e]; }
        } else {
            const float4* pa0 = (const float4*)(xrowF + k0 + laneHalf * 8);
            const float4* pa1 = (const float4*)(xrowF + k0 + 16 + laneHalf * 8);
            float4 a0 = pa0[0], a1 = pa0[1];
            float4 a2 = pa1[0], a3 = pa1[1];
            a[0]=(_Float16)a0.x; a[1]=(_Float16)a0.y; a[2]=(_Float16)a0.z; a[3]=(_Float16)a0.w;
            a[4]=(_Float16)a1.x; a[5]=(_Float16)a1.y; a[6]=(_Float16)a1.z; a[7]=(_Float16)a1.w;
            a[8]=(_Float16)a2.x; a[9]=(_Float16)a2.y; a[10]=(_Float16)a2.z; a[11]=(_Float16)a2.w;
            a[12]=(_Float16)a3.x; a[13]=(_Float16)a3.y; a[14]=(_Float16)a3.z; a[15]=(_Float16)a3.w;
        }
        // ---- B fragments: pre-packed, contiguous 32B per lane (b128 x2 each)
        v16h bfrag[4];
        #pragma unroll
        for (int t = 0; t < 4; ++t)
            bfrag[t] = *(const v16h*)(PB +
                (((size_t)(tile0 + t) * (K / 32) + kblk) * 32 + lane) * 16);
        // four independent accumulators: back-to-back WMMAs, no RAW bubbles
        #pragma unroll
        for (int t = 0; t < 4; ++t)
            acc[t] = __builtin_amdgcn_wmma_f32_16x16x32_f16(
                         false, a, false, bfrag[t], (short)0, acc[t], false, false);
    }

    #pragma unroll
    for (int t = 0; t < 4; ++t) {
        const int c = colBase + t * 16 + lane15;
        const float bs = bias[c];
        float g = 1.0f, bb = 0.0f;
        if (MODE == 1) { g = bn_g[c] * BN_INV; bb = bn_b[c]; }
        #pragma unroll
        for (int i = 0; i < 8; ++i) {
            int row = rowBase + i + laneHalf * 8;
            float v = acc[t][i] + bs;
            if (MODE == 1) v = fmaxf(v * g + bb, 0.0f);
            if (MODE == 3) v = fast_tanh(v);
            if (OUTF16) ((_Float16*)Yv)[(size_t)row * H_ + c] = (_Float16)v;
            else        ((float*)Yv)[(size_t)row * H_ + c] = v;
        }
    }
}

// ============================================================
// Node encoder: h = relu(bn(nf @ W + b)); writes h, hjk slice 0 (f32 + f16)
// ============================================================
__global__ void encode_kernel(const float* __restrict__ nf,
                              const float* __restrict__ w,   // DN x H
                              const float* __restrict__ b,
                              const float* __restrict__ bn_g,
                              const float* __restrict__ bn_b,
                              float* __restrict__ h, float* __restrict__ hjk,
                              _Float16* __restrict__ hjkh)
{
    size_t idx = (size_t)blockIdx.x * 256 + threadIdx.x;
    int c = (int)(idx & (H_ - 1));
    size_t node = idx >> 7;
    const float* x = nf + node * DN_;
    float s = b[c];
    #pragma unroll
    for (int k = 0; k < DN_; ++k) s += x[k] * w[k * H_ + c];
    s = s * (bn_g[c] * BN_INV) + bn_b[c];
    s = fmaxf(s, 0.0f);
    h[idx] = s;
    hjk[node * JK_ + c]  = s;
    hjkh[node * JK_ + c] = (_Float16)s;
}

// agg = (1+eps)*h
__global__ void initagg_kernel(const float* __restrict__ h,
                               const float* __restrict__ eps,
                               float* __restrict__ agg)
{
    size_t idx = (size_t)blockIdx.x * 256 + threadIdx.x;
    agg[idx] = (1.0f + eps[0]) * h[idx];
}

// zero-fill (for atomic accumulators)
__global__ void zero_kernel(float* __restrict__ p)
{
    p[(size_t)blockIdx.x * 256 + threadIdx.x] = 0.0f;
}

// message + scatter-add: one block (128 threads) per (b, edge)
__global__ void message_kernel(const float* __restrict__ h,
                               const int* __restrict__ eidx,
                               const int* __restrict__ etype,
                               const float* __restrict__ eemb,
                               float* __restrict__ agg)
{
    int be = blockIdx.x;              // [0, B*E)
    int b  = be / E_;
    int e  = be - b * E_;
    int c  = threadIdx.x;             // [0,128)
    int src = eidx[(size_t)b * 2 * E_ + e];
    int dst = eidx[(size_t)b * 2 * E_ + E_ + e];
    int t   = etype[(size_t)b * E_ + e];
    float m = h[((size_t)b * N_ + src) * H_ + c] + eemb[t * H_ + c];
    m = fmaxf(m, 0.0f);
    atomicAdd(&agg[((size_t)b * N_ + dst) * H_ + c], m);
}

// post-MLP: bn -> mask -> residual -> layernorm ; one block (128 thr) per node
__global__ void post_mlp_kernel(const float* __restrict__ y2,
                                const float* __restrict__ h_prev,
                                const float* __restrict__ bn_g,
                                const float* __restrict__ bn_b,
                                const float* __restrict__ ln_g,
                                const float* __restrict__ ln_b,
                                const unsigned char* __restrict__ mask,
                                float* __restrict__ h_out)
{
    size_t node = blockIdx.x;
    int c = threadIdx.x;
    float mk = (float)mask[node];
    float v = y2[node * H_ + c];
    v = (v * (bn_g[c] * BN_INV) + bn_b[c]) * mk;
    float x = h_prev[node * H_ + c] + v;

    __shared__ float red[4];
    float s = x;
    for (int o = 16; o > 0; o >>= 1) s += __shfl_down(s, o, 32);
    if ((c & 31) == 0) red[c >> 5] = s;
    __syncthreads();
    float mu = (red[0] + red[1] + red[2] + red[3]) * (1.0f / H_);
    float d = x - mu;
    float s2 = d * d;
    __syncthreads();
    for (int o = 16; o > 0; o >>= 1) s2 += __shfl_down(s2, o, 32);
    if ((c & 31) == 0) red[c >> 5] = s2;
    __syncthreads();
    float var = (red[0] + red[1] + red[2] + red[3]) * (1.0f / H_);
    h_out[node * H_ + c] = d * rsqrtf(var + 1e-5f) * ln_g[c] + ln_b[c];
}

// column-sum over nodes, coalesced: threads = channels, blocks partition nodes.
// grid (B, 32), block 128 ; partial sums -> atomicAdd into zeroed out[b,c].
__global__ void colsum_kernel(const float* __restrict__ h,
                              const unsigned char* __restrict__ mask,
                              float* __restrict__ out)
{
    int b = blockIdx.x, c = threadIdx.x;
    int n0 = blockIdx.y * (N_ / 32);
    float s = 0.0f;
    for (int n = n0; n < n0 + N_ / 32; ++n)
        s += h[((size_t)b * N_ + n) * H_ + c] * (float)mask[(size_t)b * N_ + n];
    atomicAdd(&out[b * H_ + c], s);
}

// virtual node update: grid B, block 128
__global__ void vn_kernel(const float* __restrict__ vn_in,
                          const float* __restrict__ nodesum,
                          const float* __restrict__ w1, const float* __restrict__ b1,
                          const float* __restrict__ g1, const float* __restrict__ bb1,
                          const float* __restrict__ w2, const float* __restrict__ b2,
                          const float* __restrict__ g2, const float* __restrict__ bb2,
                          float* __restrict__ vn_out)
{
    int b = blockIdx.x, c = threadIdx.x;
    __shared__ float v0[H_], y1[H_];
    v0[c] = vn_in[b * H_ + c] + nodesum[b * H_ + c];
    __syncthreads();
    float s = b1[c];
    for (int k = 0; k < H_; ++k) s += v0[k] * w1[k * H_ + c];
    s = fmaxf(s * (g1[c] * BN_INV) + bb1[c], 0.0f);
    y1[c] = s;
    __syncthreads();
    float t = b2[c];
    for (int k = 0; k < H_; ++k) t += y1[k] * w2[k * H_ + c];
    t = t * (g2[c] * BN_INV) + bb2[c] + vn_in[b * H_ + c];
    vn_out[b * H_ + c] = t;
}

// h = (h + sigmoid(gate)*vn[b]) * mask ; writes h_jk slot (f32 + f16)
__global__ void apply_vn_kernel(float* __restrict__ h,
                                const float* __restrict__ vn,
                                const float* __restrict__ gate,
                                const unsigned char* __restrict__ mask,
                                float* __restrict__ hjk,
                                _Float16* __restrict__ hjkh, int slot)
{
    size_t idx = (size_t)blockIdx.x * 256 + threadIdx.x;
    int c = (int)(idx & (H_ - 1));
    size_t node = idx >> 7;
    int b = (int)(node >> 12);      // N = 4096
    float sg = 1.0f / (1.0f + __expf(-gate[0]));
    float v = (h[idx] + sg * vn[b * H_ + c]) * (float)mask[node];
    h[idx] = v;
    hjk[node * JK_ + (size_t)slot * H_ + c]  = v;
    hjkh[node * JK_ + (size_t)slot * H_ + c] = (_Float16)v;
}

// vn init: vn[b,c] = vn_init[c]
__global__ void vninit_kernel(const float* __restrict__ vi, float* __restrict__ vn)
{
    int idx = blockIdx.x * 256 + threadIdx.x;   // B*H = 4096
    vn[idx] = vi[idx & (H_ - 1)];
}

// scores[node] = dot(tanh_proj[node,:], att_a) ; block 128 per node
__global__ void score_kernel(const float* __restrict__ t,
                             const float* __restrict__ att_a,
                             float* __restrict__ scores)
{
    size_t node = blockIdx.x;
    int c = threadIdx.x;
    float p = t[node * H_ + c] * att_a[c];
    __shared__ float red[4];
    for (int o = 16; o > 0; o >>= 1) p += __shfl_down(p, o, 32);
    if ((c & 31) == 0) red[c >> 5] = p;
    __syncthreads();
    if (c == 0) scores[node] = red[0] + red[1] + red[2] + red[3];
}

// masked softmax per batch over N ; grid B, block 256
__global__ void softmax_kernel(const float* __restrict__ scores,
                               const unsigned char* __restrict__ mask,
                               float* __restrict__ alpha)
{
    int b = blockIdx.x;
    __shared__ float sm[256];
    float mx = -3.4e38f;
    for (int n = threadIdx.x; n < N_; n += 256) {
        float s = mask[(size_t)b * N_ + n] ? scores[(size_t)b * N_ + n] : -3.4e38f;
        mx = fmaxf(mx, s);
    }
    sm[threadIdx.x] = mx; __syncthreads();
    for (int o = 128; o > 0; o >>= 1) {
        if (threadIdx.x < o) sm[threadIdx.x] = fmaxf(sm[threadIdx.x], sm[threadIdx.x + o]);
        __syncthreads();
    }
    mx = sm[0]; __syncthreads();
    float sum = 0.0f;
    for (int n = threadIdx.x; n < N_; n += 256) {
        float s = mask[(size_t)b * N_ + n] ? scores[(size_t)b * N_ + n] : -3.4e38f;
        sum += __expf(s - mx);
    }
    sm[threadIdx.x] = sum; __syncthreads();
    for (int o = 128; o > 0; o >>= 1) {
        if (threadIdx.x < o) sm[threadIdx.x] += sm[threadIdx.x + o];
        __syncthreads();
    }
    float inv = 1.0f / sm[0];
    for (int n = threadIdx.x; n < N_; n += 256) {
        unsigned char m = mask[(size_t)b * N_ + n];
        float s = m ? scores[(size_t)b * N_ + n] : -3.4e38f;
        alpha[(size_t)b * N_ + n] = __expf(s - mx) * inv * (m ? 1.0f : 0.0f);
    }
}

// pool partials, coalesced: threads = 128 consecutive j, blocks partition nodes.
// grid (B, JK/128=6, 4) block 128 ; atomicAdd partials into zeroed accumulators.
__global__ void pool_partial_kernel(const float* __restrict__ hjk,
                                    const float* __restrict__ alpha,
                                    const unsigned char* __restrict__ mask,
                                    float* __restrict__ attn_acc,
                                    float* __restrict__ sum_acc)
{
    int b = blockIdx.x;
    int j = blockIdx.y * 128 + threadIdx.x;
    int n0 = blockIdx.z * (N_ / 4);
    float sa = 0.0f, ss = 0.0f;
    for (int n = n0; n < n0 + N_ / 4; ++n) {
        float v = hjk[((size_t)b * N_ + n) * JK_ + j];
        sa += alpha[(size_t)b * N_ + n] * v;
        ss += mask[(size_t)b * N_ + n] ? v : 0.0f;
    }
    atomicAdd(&attn_acc[(size_t)b * JK_ + j], sa);
    atomicAdd(&sum_acc[(size_t)b * JK_ + j], ss);
}

// graw = g*attn_pool + (1-g)*sum_pool
__global__ void pool_final_kernel(const float* __restrict__ attn_acc,
                                  const float* __restrict__ sum_acc,
                                  const float* __restrict__ gate,
                                  float* __restrict__ graw)
{
    size_t idx = (size_t)blockIdx.x * 256 + threadIdx.x;   // B*JK
    float g = 1.0f / (1.0f + __expf(-gate[0]));
    graw[idx] = g * attn_acc[idx] + (1.0f - g) * sum_acc[idx];
}

// small head GEMM: Y[b*ldY + o] = epi(X[b,:K]@W[K,O] + bias) ; grid B, 256 thr
__global__ void head_kernel(const float* __restrict__ X, int ldX,
                            const float* __restrict__ w, const float* __restrict__ bias,
                            const float* __restrict__ bn_g, const float* __restrict__ bn_b,
                            float* __restrict__ Y, int ldY, int K, int O, int mode)
{
    __shared__ float xs[JK_];
    int b = blockIdx.x;
    for (int k = threadIdx.x; k < K; k += blockDim.x) xs[k] = X[(size_t)b * ldX + k];
    __syncthreads();
    for (int o = threadIdx.x; o < O; o += blockDim.x) {
        float s = bias[o];
        for (int k = 0; k < K; ++k) s += xs[k] * w[(size_t)k * O + o];
        if (mode >= 1) s = s * (bn_g[o] * BN_INV) + bn_b[o];
        if (mode == 1) s = fmaxf(s, 0.0f);
        Y[(size_t)b * ldY + o] = s;
    }
}

// ---------------- param indices (JAX sorted-key pytree flattening) ----------
enum {
    IN_NF = 0, IN_EIDX, IN_ETYPE, IN_MASK, IN_HC,
    P_ATTW_B = 5, P_ATTW_W, P_ATTA, P_ATTGATE,
    P_CLSBN_B, P_CLSBN_G, P_CLS1_B, P_CLS1_W, P_CLS2_B, P_CLS2_W,
    P_EDGEEMB,
    P_FBN1_B, P_FBN1_G, P_FBN2_B, P_FBN2_G,
    P_FL1_B, P_FL1_W, P_FL2_B, P_FL2_W,
    P_GBN_B, P_GBN_G, P_GL_B, P_GL_W,
    P_LAYER0 = 28,     // 20 leaves per layer
    P_NEBN_B = 128, P_NEBN_G, P_NEL_B, P_NEL_W, P_VNINIT
};
// per-layer leaf offsets (sorted keys):
#define L_BN_B   0
#define L_BN_G   1
#define L_EPS    2
#define L_LN_B   3
#define L_LN_G   4
#define L_MBN_B  5
#define L_MBN_G  6
#define L_M1_B   7
#define L_M1_W   8
#define L_M2_B   9
#define L_M2_W   10
#define L_VBN1_B 11
#define L_VBN1_G 12
#define L_VBN2_B 13
#define L_VBN2_G 14
#define L_VGATE  15
#define L_V1_B   16
#define L_V1_W   17
#define L_V2_B   18
#define L_V2_W   19

extern "C" void kernel_launch(void* const* d_in, const int* in_sizes, int n_in,
                              void* d_out, int out_size, void* d_ws, size_t ws_size,
                              hipStream_t stream)
{
    if (n_in < 133) return;
    auto F  = [&](int i) { return (const float*)d_in[i]; };
    auto LP = [&](int li, int off) { return (const float*)d_in[P_LAYER0 + 20 * li + off]; };

    const float* nf    = F(IN_NF);
    const int*   eidx  = (const int*)d_in[IN_EIDX];
    const int*   etyp  = (const int*)d_in[IN_ETYPE];
    const unsigned char* mask = (const unsigned char*)d_in[IN_MASK];
    const float* hand  = F(IN_HC);

    // workspace bump allocation (256B-aligned chunks)
    char* ws = (char*)d_ws;
    size_t off = 0;
    auto allocF = [&](size_t n) { float* p = (float*)(ws + off);
                                  off += ((n * 4 + 255) / 256) * 256; return p; };
    auto allocH = [&](size_t n) { _Float16* p = (_Float16*)(ws + off);
                                  off += ((n * 2 + 255) / 256) * 256; return p; };
    const size_t MH = (size_t)B_ * N_ * H_;     // 16.7M
    float*    h     = allocF(MH);
    float*    agg   = allocF(MH);
    _Float16* t1h   = allocH(MH);
    float*    t2    = allocF(MH);
    float*    hjk   = allocF((size_t)B_ * N_ * JK_);
    _Float16* hjkh  = allocH((size_t)B_ * N_ * JK_);
    _Float16* pw1   = allocH((size_t)H_ * H_);
    _Float16* pw2   = allocH((size_t)H_ * H_);
    _Float16* pattw = allocH((size_t)JK_ * H_);
    float* vnA      = allocF(B_ * H_);
    float* vnB      = allocF(B_ * H_);
    float* nodesum  = allocF(B_ * H_);
    float* scores   = allocF((size_t)B_ * N_);
    float* alpha    = allocF((size_t)B_ * N_);
    float* attn_acc = allocF((size_t)B_ * JK_);
    float* sum_acc  = allocF((size_t)B_ * JK_);
    float* graw     = allocF((size_t)B_ * JK_);
    float* f1       = allocF(B_ * FUS_);
    float* comb     = allocF(B_ * 2 * FUS_);
    float* c1       = allocF(B_ * H_);
    (void)ws_size;

    const int M = B_ * N_;                     // 131072
    dim3 gemmGrid(M / 128, H_ / 64);           // (1024, 2)
    int eltBlocks = (int)(MH / 256);           // 65536

    // encoder -> h, hjk[:,0:H] (f32+f16)
    encode_kernel<<<eltBlocks, 256, 0, stream>>>(
        nf, F(P_NEL_W), F(P_NEL_B), F(P_NEBN_G), F(P_NEBN_B), h, hjk, hjkh);
    // vn = vn_init
    vninit_kernel<<<(B_ * H_) / 256, 256, 0, stream>>>(F(P_VNINIT), vnA);

    float* vn_cur = vnA;
    float* vn_nxt = vnB;

    for (int li = 0; li < L_; ++li) {
        // pack this layer's MLP weights into WMMA fragment order (f16)
        pack_w_kernel<H_><<<(H_ * H_) / 256, 256, 0, stream>>>(LP(li, L_M1_W), pw1);
        pack_w_kernel<H_><<<(H_ * H_) / 256, 256, 0, stream>>>(LP(li, L_M2_W), pw2);
        // agg = (1+eps)*h ; scatter messages
        initagg_kernel<<<eltBlocks, 256, 0, stream>>>(h, LP(li, L_EPS), agg);
        message_kernel<<<B_ * E_, H_, 0, stream>>>(h, eidx, etyp, F(P_EDGEEMB), agg);
        // t1h = relu(bn(agg @ W1 + b1))  (f32 in, f16 out)
        wmma_gemm<H_, 1, false, true><<<gemmGrid, 256, 0, stream>>>(
            agg, pw1, LP(li, L_M1_B), LP(li, L_MBN_G), LP(li, L_MBN_B), t1h);
        // t2 = t1h @ W2 + b2  (f16 in, f32 out)
        wmma_gemm<H_, 0, true, false><<<gemmGrid, 256, 0, stream>>>(
            t1h, pw2, LP(li, L_M2_B), nullptr, nullptr, t2);
        // bn, mask, residual, layernorm -> h (in place)
        post_mlp_kernel<<<M, H_, 0, stream>>>(
            t2, h, LP(li, L_BN_G), LP(li, L_BN_B), LP(li, L_LN_G), LP(li, L_LN_B),
            mask, h);
        // node_sum (coalesced partials) and virtual node
        zero_kernel<<<(B_ * H_) / 256, 256, 0, stream>>>(nodesum);
        colsum_kernel<<<dim3(B_, 32), H_, 0, stream>>>(h, mask, nodesum);
        vn_kernel<<<B_, H_, 0, stream>>>(
            vn_cur, nodesum,
            LP(li, L_V1_W), LP(li, L_V1_B), LP(li, L_VBN1_G), LP(li, L_VBN1_B),
            LP(li, L_V2_W), LP(li, L_V2_B), LP(li, L_VBN2_G), LP(li, L_VBN2_B),
            vn_nxt);
        // gate-broadcast, mask, write h_jk slot li+1 (f32+f16)
        apply_vn_kernel<<<eltBlocks, 256, 0, stream>>>(
            h, vn_nxt, LP(li, L_VGATE), mask, hjk, hjkh, li + 1);
        float* tmp = vn_cur; vn_cur = vn_nxt; vn_nxt = tmp;
    }

    // attention: t2 = tanh(hjkh @ att_W + b)  (K=768, f16 in, f32 out)
    pack_w_kernel<JK_><<<(JK_ * H_) / 256, 256, 0, stream>>>(F(P_ATTW_W), pattw);
    wmma_gemm<JK_, 3, true, false><<<gemmGrid, 256, 0, stream>>>(
        hjkh, pattw, F(P_ATTW_B), nullptr, nullptr, t2);
    score_kernel<<<M, H_, 0, stream>>>(t2, F(P_ATTA), scores);
    softmax_kernel<<<B_, 256, 0, stream>>>(scores, mask, alpha);
    // pooling: coalesced partials + finalize
    zero_kernel<<<(B_ * JK_) / 256, 256, 0, stream>>>(attn_acc);
    zero_kernel<<<(B_ * JK_) / 256, 256, 0, stream>>>(sum_acc);
    pool_partial_kernel<<<dim3(B_, JK_ / 128, 4), 128, 0, stream>>>(
        hjk, alpha, mask, attn_acc, sum_acc);
    pool_final_kernel<<<(B_ * JK_) / 256, 256, 0, stream>>>(
        attn_acc, sum_acc, F(P_ATTGATE), graw);

    // heads: graph -> comb[:,0:256]
    head_kernel<<<B_, 256, 0, stream>>>(graw, JK_, F(P_GL_W), F(P_GL_B),
        F(P_GBN_G), F(P_GBN_B), comb, 2 * FUS_, JK_, FUS_, 1);
    // f1 = relu(bn(hand @ feat1))
    head_kernel<<<B_, 256, 0, stream>>>(hand, HC_, F(P_FL1_W), F(P_FL1_B),
        F(P_FBN1_G), F(P_FBN1_B), f1, FUS_, HC_, FUS_, 1);
    // f2 -> comb[:,256:512]
    head_kernel<<<B_, 256, 0, stream>>>(f1, FUS_, F(P_FL2_W), F(P_FL2_B),
        F(P_FBN2_G), F(P_FBN2_B), comb + FUS_, 2 * FUS_, FUS_, FUS_, 1);
    // c1 = relu(bn(comb @ cls1))
    head_kernel<<<B_, 256, 0, stream>>>(comb, 2 * FUS_, F(P_CLS1_W), F(P_CLS1_B),
        F(P_CLSBN_G), F(P_CLSBN_B), c1, H_, 2 * FUS_, H_, 1);
    // logits -> d_out
    head_kernel<<<B_, 256, 0, stream>>>(c1, H_, F(P_CLS2_W), F(P_CLS2_B),
        nullptr, nullptr, (float*)d_out, NC_, H_, NC_, 0);
    (void)in_sizes; (void)out_size;
}
